// LSTMRegressor_82781199663183
// MI455X (gfx1250) — matrix operand order repaired and checked by reading, not compiled, not run on hardware
//
#include <hip/hip_runtime.h>

// ---------------------------------------------------------------------------
// 2-layer LSTM (H=1024) + linear head, bf16 WMMA (v_wmma_f32_16x16x32_bf16)
// B=128, T=512, I=64, O=64.
//
// MI455X reasoning:
//  * Weights ~25MB bf16 -> fully L2-resident (192MB). GEMMs are matrix-pipe
//    bound, so bf16 WMMA with fp32 accumulate; fp32 cell state + activations.
//  * Per-step kernels are ~1-3us of math => schedule is launch-latency bound.
//    So the O=64 head GEMM for step t-1 is fused into layer-1 of step t
//    (valid: h2 is double-buffered, the read buffer is never written that
//    step). Graph nodes: 1536 -> 1026.
//  * h buffers double-buffered to avoid cross-block RAW on the recurrence;
//    c updated in place (block-exclusive columns).
// ---------------------------------------------------------------------------

typedef __bf16 bf16;
typedef __attribute__((ext_vector_type(16))) __bf16 v16bf;
typedef __attribute__((ext_vector_type(8)))  __bf16 v8bf;
typedef __attribute__((ext_vector_type(8)))  float  v8f;

#define BATCH 128
#define SEQ   512
#define IN    64
#define HID   1024
#define OUT   64
#define GATES 4096  // 4*HID

// ---- WMMA fragment loaders (wave32, contiguous k order shared by A and B) ---------------------

// A-matrix 16x32 bf16 (row-major source, leading dim lda):
// lanes 0-15: row M=lane, K = kb+0..15
// lanes 16-31: row M=lane-16, K = kb+16..31
__device__ __forceinline__ v16bf load_a_frag(const bf16* __restrict__ A,
                                             int lda, int m0, int kb, int lane) {
  const int half = lane >> 4;
  const int m    = m0 + (lane & 15);
  const bf16* p  = A + (size_t)m * lda + kb + half * 16;
  union { v16bf v; v8bf h[2]; } u;
  u.h[0] = *(const v8bf*)(p);        // K offsets +0..7 (of this half)
  u.h[1] = *(const v8bf*)(p + 8);    // K offsets +8..15 (of this half)
  return u.v;
}

// B-matrix 32x16 bf16 where B[k][n] = W[n][k] (W row-major [N,K]):
// lane holds column n = n0 + (lane&15); lanes 0-15: K=kb..kb+15,
// lanes 16-31: K=kb+16..kb+31. 16 contiguous bf16 per lane = 32B coalesced.
__device__ __forceinline__ v16bf load_b_frag(const bf16* __restrict__ Wrow,
                                             int kb, int lane) {
  return *(const v16bf*)(Wrow + kb + (lane >> 4) * 16);
}

__device__ __forceinline__ v8f wmma_bf16(v16bf a, v16bf b, v8f c) {
  return __builtin_amdgcn_wmma_f32_16x16x32_bf16(
      /*neg_a=*/false, a, /*neg_b=*/false, b,
      /*c_mod=*/(short)0, c, /*reuse_a=*/false, /*reuse_b=*/false);
}

__device__ __forceinline__ float sigmoidf_(float x) {
  return 1.0f / (1.0f + __expf(-x));
}

// ---- prep: fp32 -> bf16 conversion, bias fusion, state init ----------------
__global__ void lstm_prep_kernel(
    const float* __restrict__ Wih1, const float* __restrict__ Whh1,
    const float* __restrict__ bih1, const float* __restrict__ bhh1,
    const float* __restrict__ Wih2, const float* __restrict__ Whh2,
    const float* __restrict__ bih2, const float* __restrict__ bhh2,
    const float* __restrict__ Wout, const float* __restrict__ x,
    bf16* __restrict__ Wih1b, bf16* __restrict__ Whh1b,
    bf16* __restrict__ Wih2b, bf16* __restrict__ Whh2b,
    bf16* __restrict__ Woutb, bf16* __restrict__ xb,
    float* __restrict__ b1s, float* __restrict__ b2s,
    float* __restrict__ c1, float* __restrict__ c2,
    bf16* __restrict__ h1b0, bf16* __restrict__ h2b0) {
  const size_t i  = (size_t)blockIdx.x * blockDim.x + threadIdx.x;
  const size_t st = (size_t)gridDim.x * blockDim.x;
  for (size_t j = i; j < (size_t)GATES * IN;  j += st) Wih1b[j] = (bf16)Wih1[j];
  for (size_t j = i; j < (size_t)GATES * HID; j += st) Whh1b[j] = (bf16)Whh1[j];
  for (size_t j = i; j < (size_t)GATES * HID; j += st) Wih2b[j] = (bf16)Wih2[j];
  for (size_t j = i; j < (size_t)GATES * HID; j += st) Whh2b[j] = (bf16)Whh2[j];
  for (size_t j = i; j < (size_t)OUT * HID;   j += st) Woutb[j] = (bf16)Wout[j];
  for (size_t j = i; j < (size_t)BATCH * SEQ * IN; j += st) xb[j] = (bf16)x[j];
  for (size_t j = i; j < (size_t)GATES; j += st) {
    b1s[j] = bih1[j] + bhh1[j];
    b2s[j] = bih2[j] + bhh2[j];
  }
  for (size_t j = i; j < (size_t)BATCH * HID; j += st) {
    c1[j] = 0.0f; c2[j] = 0.0f;
    h1b0[j] = (bf16)0.0f; h2b0[j] = (bf16)0.0f;
  }
}

// ---- one LSTM layer, one timestep (+ fused head GEMM of previous step) -----
// gates = A0 @ W0^T + A1 @ W1^T + bsum ;  (i,f,g,o) -> c,h update
// Grid: 64 blocks x 256 threads. Block nb owns h columns [nb*16, nb*16+16).
// Wave = (gate 0..3) x (batch half 0..1): 64x16 fp32 tile, 4 v8f accumulators.
// If yt >= 0: blocks 0..31, wave 7 additionally compute one 16x16 tile of
// y_yt = h2y @ Wout^T + bout (h2y is read-stable this step).
__global__ __launch_bounds__(256) void lstm_layer_kernel(
    const bf16* __restrict__ A0, int lda0, int K0,          // input part
    const bf16* __restrict__ W0, int ldw0,                  // W_ih [4096,K0]
    const bf16* __restrict__ A1,                            // h_prev [128,1024]
    const bf16* __restrict__ W1,                            // W_hh [4096,1024]
    const float* __restrict__ bsum,                         // [4096]
    float* __restrict__ c,                                  // [128,1024] inout
    bf16* __restrict__ h_out,                               // [128,1024]
    const bf16* __restrict__ h2y,                           // h2 for fused head
    const bf16* __restrict__ Wouty,                         // [64,1024]
    const float* __restrict__ bouty,                        // [64]
    float* __restrict__ yout,                               // [128,512,64]
    int yt) {
  __shared__ float sg[4][BATCH][16];  // 32KB gate exchange

  const int tid   = threadIdx.x;
  const int lane  = tid & 31;
  const int wave  = tid >> 5;
  const int gate  = wave & 3;
  const int mhalf = wave >> 2;
  const int m0    = mhalf * 64;
  const int ncol  = blockIdx.x * 16;                 // h-column base
  const int nlane = gate * HID + ncol + (lane & 15); // gate-column for B lane

  v8f acc[4] = {};

  // input contribution
  {
    const bf16* Wrow = W0 + (size_t)nlane * ldw0;
    for (int kb = 0; kb < K0; kb += 32) {
      v16bf bf = load_b_frag(Wrow, kb, lane);
#pragma unroll
      for (int mt = 0; mt < 4; ++mt) {
        v16bf af = load_a_frag(A0, lda0, m0 + mt * 16, kb, lane);
        acc[mt] = wmma_bf16(af, bf, acc[mt]);
      }
    }
  }
  // recurrent contribution
  {
    const bf16* Wrow = W1 + (size_t)nlane * HID;
    for (int kb = 0; kb < HID; kb += 32) {
      v16bf bf = load_b_frag(Wrow, kb, lane);
#pragma unroll
      for (int mt = 0; mt < 4; ++mt) {
        v16bf af = load_a_frag(A1, HID, m0 + mt * 16, kb, lane);
        acc[mt] = wmma_bf16(af, bf, acc[mt]);
      }
    }
  }

  // D layout: vgpr r, lane l -> row = r + 8*(l>=16), col = l&15
  const int dcol   = lane & 15;
  const int rowadd = (lane >> 4) * 8;
#pragma unroll
  for (int mt = 0; mt < 4; ++mt)
#pragma unroll
    for (int r = 0; r < 8; ++r)
      sg[gate][m0 + mt * 16 + rowadd + r][dcol] = acc[mt][r];
  __syncthreads();

  // elementwise cell update: 128 rows x 16 cols
#pragma unroll
  for (int it = 0; it < (BATCH * 16) / 256; ++it) {
    const int idx = tid + it * 256;
    const int row = idx >> 4;
    const int col = idx & 15;
    const int n   = ncol + col;
    const float ig = sigmoidf_(sg[0][row][col] + bsum[n]);
    const float fg = sigmoidf_(sg[1][row][col] + bsum[HID + n]);
    const float gg = tanhf    (sg[2][row][col] + bsum[2 * HID + n]);
    const float og = sigmoidf_(sg[3][row][col] + bsum[3 * HID + n]);
    const size_t ci = (size_t)row * HID + n;
    const float cn = fg * c[ci] + ig * gg;
    c[ci] = cn;
    h_out[ci] = (bf16)(og * tanhf(cn));
  }

  // fused head: one 16x16 tile of y_yt per block (blocks 0..31), wave 7
  if (yt >= 0 && blockIdx.x < (BATCH / 16) * (OUT / 16) && wave == 7) {
    const int m0y = (blockIdx.x >> 2) * 16;  // 8 M-tiles
    const int n0y = (blockIdx.x & 3) * 16;   // 4 N-tiles
    const bf16* Wrow = Wouty + (size_t)(n0y + (lane & 15)) * HID;
    v8f yacc = {};
    for (int kb = 0; kb < HID; kb += 32) {
      v16bf bf = load_b_frag(Wrow, kb, lane);
      v16bf af = load_a_frag(h2y, HID, m0y, kb, lane);
      yacc = wmma_bf16(af, bf, yacc);
    }
    const float bias = bouty[n0y + dcol];
#pragma unroll
    for (int r = 0; r < 8; ++r) {
      const int m = m0y + rowadd + r;
      yout[(size_t)m * SEQ * OUT + (size_t)yt * OUT + n0y + dcol] = yacc[r] + bias;
    }
  }
}

// ---- standalone head (used for the final timestep only) --------------------
__global__ __launch_bounds__(32) void lstm_out_kernel(
    const bf16* __restrict__ h2, const bf16* __restrict__ Wout,
    const float* __restrict__ bout, float* __restrict__ out, int t) {
  const int lane = threadIdx.x;
  const int m0 = blockIdx.x * 16;
  const int n0 = blockIdx.y * 16;
  const bf16* Wrow = Wout + (size_t)(n0 + (lane & 15)) * HID;

  v8f acc = {};
  for (int kb = 0; kb < HID; kb += 32) {
    v16bf bf = load_b_frag(Wrow, kb, lane);
    v16bf af = load_a_frag(h2, HID, m0, kb, lane);
    acc = wmma_bf16(af, bf, acc);
  }

  const int col    = lane & 15;
  const int rowadd = (lane >> 4) * 8;
  const float bias = bout[n0 + col];
#pragma unroll
  for (int r = 0; r < 8; ++r) {
    const int m = m0 + rowadd + r;
    out[(size_t)m * SEQ * OUT + (size_t)t * OUT + n0 + col] = acc[r] + bias;
  }
}

// ---------------------------------------------------------------------------
extern "C" void kernel_launch(void* const* d_in, const int* in_sizes, int n_in,
                              void* d_out, int out_size, void* d_ws, size_t ws_size,
                              hipStream_t stream) {
  const float* x    = (const float*)d_in[0];
  const float* Wih1 = (const float*)d_in[1];
  const float* Whh1 = (const float*)d_in[2];
  const float* bih1 = (const float*)d_in[3];
  const float* bhh1 = (const float*)d_in[4];
  const float* Wih2 = (const float*)d_in[5];
  const float* Whh2 = (const float*)d_in[6];
  const float* bih2 = (const float*)d_in[7];
  const float* bhh2 = (const float*)d_in[8];
  const float* Wout = (const float*)d_in[9];
  const float* bout = (const float*)d_in[10];
  float* out = (float*)d_out;

  // workspace carve-up (256B aligned)
  char* ws = (char*)d_ws;
  size_t off = 0;
  auto carve = [&](size_t bytes) {
    void* p = ws + off;
    off = (off + bytes + 255) & ~(size_t)255;
    return p;
  };
  bf16* Wih1b = (bf16*)carve((size_t)GATES * IN  * 2);
  bf16* Whh1b = (bf16*)carve((size_t)GATES * HID * 2);
  bf16* Wih2b = (bf16*)carve((size_t)GATES * HID * 2);
  bf16* Whh2b = (bf16*)carve((size_t)GATES * HID * 2);
  bf16* Woutb = (bf16*)carve((size_t)OUT * HID * 2);
  bf16* xb    = (bf16*)carve((size_t)BATCH * SEQ * IN * 2);
  float* b1s  = (float*)carve((size_t)GATES * 4);
  float* b2s  = (float*)carve((size_t)GATES * 4);
  float* c1   = (float*)carve((size_t)BATCH * HID * 4);
  float* c2   = (float*)carve((size_t)BATCH * HID * 4);
  bf16* h1b[2] = { (bf16*)carve((size_t)BATCH * HID * 2),
                   (bf16*)carve((size_t)BATCH * HID * 2) };
  bf16* h2b[2] = { (bf16*)carve((size_t)BATCH * HID * 2),
                   (bf16*)carve((size_t)BATCH * HID * 2) };

  lstm_prep_kernel<<<2048, 256, 0, stream>>>(
      Wih1, Whh1, bih1, bhh1, Wih2, Whh2, bih2, bhh2, Wout, x,
      Wih1b, Whh1b, Wih2b, Whh2b, Woutb, xb,
      b1s, b2s, c1, c2, h1b[0], h2b[0]);

  for (int t = 0; t < SEQ; ++t) {
    const int cur = t & 1, nxt = (t + 1) & 1;
    // layer 1: x_t [128,64] (strided into xb) + h1_prev; fused y_{t-1} head
    // (reads h2b[cur], which layer 2 below does NOT write this step).
    lstm_layer_kernel<<<HID / 16, 256, 0, stream>>>(
        xb + (size_t)t * IN, SEQ * IN, IN, Wih1b, IN,
        h1b[cur], Whh1b, b1s, c1, h1b[nxt],
        h2b[cur], Woutb, bout, out, t - 1);
    // layer 2: h1_new + h2_prev
    lstm_layer_kernel<<<HID / 16, 256, 0, stream>>>(
        h1b[nxt], HID, HID, Wih2b, HID,
        h2b[cur], Whh2b, b2s, c2, h2b[nxt],
        (const bf16*)nullptr, (const bf16*)nullptr, (const float*)nullptr,
        (float*)nullptr, -1);
    // final head for t = SEQ-1
    if (t == SEQ - 1) {
      lstm_out_kernel<<<dim3(BATCH / 16, OUT / 16), 32, 0, stream>>>(
          h2b[nxt], Woutb, bout, out, t);
    }
  }
}
